// BiDirectionalTGFI_42752104465021
// MI455X (gfx1250) — matrix operand order repaired and checked by reading, not compiled, not run on hardware
//
#include <hip/hip_runtime.h>
#include <hip/hip_bf16.h>
#include <math.h>

// ---------------------------------------------------------------------------
// Problem constants (match reference)
// ---------------------------------------------------------------------------
#define BB 32      // batch
#define CC 256     // channels
#define HH 80
#define WW 80
#define NN 6400    // tokens = H*W
#define KK 512     // selected tokens (256 high + 256 low)
#define NH 8       // heads
#define HD 32      // head dim
#define C3 768     // 3*C

typedef _Float16 half_t;
typedef _Float16 v16h __attribute__((ext_vector_type(16)));
typedef _Float16 v8h  __attribute__((ext_vector_type(8)));
typedef _Float16 v4h  __attribute__((ext_vector_type(4)));
typedef float    v8f  __attribute__((ext_vector_type(8)));
typedef float    v4f  __attribute__((ext_vector_type(4)));

union V16U { v16h v; v8h h[2]; };

// 16 contiguous halves
__device__ inline v16h ld16(const half_t* p) {
  V16U u;
  u.h[0] = *(const v8h*)p;
  u.h[1] = *(const v8h*)(p + 8);
  return u.v;
}
// A fragment per-lane data: 8 halves at p (K=base..base+7), 8 at p+16 (K=base+16..)
__device__ inline v16h ldA(const half_t* p) {
  V16U u;
  u.h[0] = *(const v8h*)p;
  u.h[1] = *(const v8h*)(p + 16);
  return u.v;
}

__device__ inline v8f wmma_f16(v16h a, v16h b, v8f c) {
  // D = A(16x32 f16) * B(32x16 f16) + C(16x16 f32)
  return __builtin_amdgcn_wmma_f32_16x16x32_f16(false, a, false, b,
                                                (short)0, c, false, false);
}

// ---------------------------------------------------------------------------
// 1. zero output
// ---------------------------------------------------------------------------
__global__ void zero_kernel(float* __restrict__ out, int n4) {
  int i = blockIdx.x * blockDim.x + threadIdx.x;
  v4f z = {0.f, 0.f, 0.f, 0.f};
  for (; i < n4; i += gridDim.x * blockDim.x)
    ((v4f*)out)[i] = z;
}

// ---------------------------------------------------------------------------
// 2. per-token L2 norm: norms[b][n] = ||x[b,:,n]||
// ---------------------------------------------------------------------------
__global__ void norms_kernel(const float* __restrict__ x,
                             float* __restrict__ norms) {
  int b = blockIdx.y;
  int n = blockIdx.x * 256 + threadIdx.x;
  const float* px = x + (size_t)b * CC * NN + n;
  float s = 0.f;
#pragma unroll 4
  for (int c = 0; c < CC; ++c) {
    float v = px[(size_t)c * NN];
    s = fmaf(v, v, s);
  }
  norms[b * NN + n] = sqrtf(s);
}

// ---------------------------------------------------------------------------
// 3. rank-based top-k (256 largest desc, then 256 smallest asc), exact
//    reproduction of jax.lax.top_k tie ordering (earlier index wins).
// ---------------------------------------------------------------------------
__global__ void topk_kernel(const float* __restrict__ norms,
                            int* __restrict__ all_idx) {
  __shared__ float sn[NN];  // 25.6 KB
  int b = blockIdx.y;
  for (int i = threadIdx.x; i < NN; i += 256)
    sn[i] = norms[b * NN + i];
  __syncthreads();
  int n = blockIdx.x * 256 + threadIdx.x;
  float my = sn[n];
  int ch = 0, cl = 0;
  for (int j = 0; j < NN; ++j) {
    float v = sn[j];
    int eq = (v == my) & (j < n);
    ch += (v > my) | eq;   // rank in descending order
    cl += (v < my) | eq;   // rank in ascending order
  }
  if (ch < 256) all_idx[b * KK + ch] = n;
  if (cl < 256) all_idx[b * KK + 256 + cl] = n;
}

// ---------------------------------------------------------------------------
// 4. weight f16 convert, layout kept:  WqkvT[col][c], WprojT[col][c]
//    (k contiguous per col row; B-fragment lanes load two 8-half runs)
// ---------------------------------------------------------------------------
__global__ void convw_kernel(const float* __restrict__ Wqkv,
                             const float* __restrict__ Wproj,
                             half_t* __restrict__ WqkvT,
                             half_t* __restrict__ WprojT) {
  int i = blockIdx.x * 256 + threadIdx.x;
  if (i < C3 * CC) {
    int col = i / CC, c = i % CC;
    WqkvT[col * CC + c] = (half_t)Wqkv[i];
  } else {
    int j = i - C3 * CC;
    int col = j / CC, c = j % CC;
    WprojT[col * CC + c] = (half_t)Wproj[j];
  }
}

// ---------------------------------------------------------------------------
// 5. gather selected tokens -> Xsel[b][kk][c] (f16, row-major, GEMM A layout)
// ---------------------------------------------------------------------------
__global__ void gather_kernel(const float* __restrict__ x,
                              const int* __restrict__ all_idx,
                              half_t* __restrict__ Xsel) {
  int b = blockIdx.y;
  int c0 = blockIdx.x * 4;
  int kk = threadIdx.x;  // 0..511
  int idx = all_idx[b * KK + kk];
  const float* px = x + (size_t)b * CC * NN + (size_t)c0 * NN + idx;
  v4h v;
  v.x = (half_t)px[0];
  v.y = (half_t)px[NN];
  v.z = (half_t)px[2 * NN];
  v.w = (half_t)px[3 * NN];
  *(v4h*)(Xsel + ((size_t)(b * KK + kk)) * CC + c0) = v;
}

// ---------------------------------------------------------------------------
// 6. QKV GEMM:  QKV[b][m][col] = sum_c Xsel[b][m][c] * WqkvT[col][c]
//    one wave per 32(m) x 64(col) tile, K-loop over 256 in steps of 32
// ---------------------------------------------------------------------------
__global__ void __launch_bounds__(32)
qkv_gemm_kernel(const half_t* __restrict__ Xsel,
                const half_t* __restrict__ WqkvT,
                half_t* __restrict__ QKV) {
  int b = blockIdx.z;
  int m0 = blockIdx.y * 32;
  int n0 = blockIdx.x * 64;
  int lane = threadIdx.x;
  int lm = lane & 15, hi = lane >> 4;

  v8f acc[2][4] = {};
  const half_t* A0 = Xsel + ((size_t)(b * KK + m0 + lm)) * CC + hi * 8;
  const half_t* A1 = A0 + 16 * CC;

  for (int c0 = 0; c0 < CC; c0 += 32) {
    if (c0 + 32 < CC) __builtin_prefetch(A0 + c0 + 32, 0, 0);
    v16h a0 = ldA(A0 + c0);
    v16h a1 = ldA(A1 + c0);
    const half_t* Bp = WqkvT + (size_t)(n0 + lm) * CC + c0 + hi * 8;
#pragma unroll
    for (int j = 0; j < 4; ++j) {
      v16h bf = ldA(Bp + j * 16 * CC);
      acc[0][j] = wmma_f16(a0, bf, acc[0][j]);
      acc[1][j] = wmma_f16(a1, bf, acc[1][j]);
    }
  }
#pragma unroll
  for (int s = 0; s < 2; ++s)
#pragma unroll
    for (int j = 0; j < 4; ++j)
#pragma unroll
      for (int g = 0; g < 8; ++g) {
        int m = m0 + s * 16 + g + hi * 8;
        int n = n0 + j * 16 + lm;
        QKV[((size_t)(b * KK + m)) * C3 + n] = (half_t)acc[s][j][g];
      }
}

// ---------------------------------------------------------------------------
// 7. attention per (b, head): out = softmax(q k^T / sqrt(hd)) @ v
//    K (key-major) and V^T (d-major, padded) staged in LDS; scores (fp32) in per-wave LDS rows;
//    two-pass softmax with wave shuffles; P (f16) overlaid into score region.
// ---------------------------------------------------------------------------
#define KT_STRIDE 520   // halves per V^T row in Vs (512 + 8 pad)
#define V_STRIDE  40    // only sizes the Vs region (KK x 40 halves)
#define S_STRIDE  516   // floats per score row (512 + 4 pad)
#define P_STRIDE  520   // halves per P row
#define ATTN_LDS_BYTES (32 * KT_STRIDE * 2 + KK * V_STRIDE * 2 + 4 * 16 * S_STRIDE * 4)

__global__ void __launch_bounds__(128)
attn_kernel(const half_t* __restrict__ QKV, half_t* __restrict__ Osel) {
  extern __shared__ char smem[];
  int h = blockIdx.x, b = blockIdx.y;
  half_t* Kt = (half_t*)smem;                       // [KK][HD], key-major
  half_t* Vs = Kt + 32 * KT_STRIDE;                 // [HD][KT_STRIDE], d-major (V^T)
  float*  Sb = (float*)(Vs + KK * V_STRIDE);        // 4 waves x [16][S_STRIDE]

  int tid = threadIdx.x, lane = tid & 31, wave = tid >> 5;
  int lm = lane & 15, hi = lane >> 4;
  float*  Sw = Sb + wave * 16 * S_STRIDE;
  half_t* Pw = (half_t*)Sw;                         // overlay (row-safe)

  const size_t qb = (size_t)b * KK * C3;
  // stage K (key-major) and V^T (d-major)
  for (int i = tid; i < KK * HD; i += 128) {
    int key = i >> 5, d = i & 31;
    const half_t* src = QKV + qb + (size_t)key * C3 + h * HD + d;
    Kt[key * HD + d] = src[CC];        // K at col offset 256
    Vs[d * KT_STRIDE + key]  = src[2 * CC];    // V at col offset 512
  }
  __syncthreads();

  const half_t hscale = (half_t)0.17677669529663687f;  // 32^-0.5

  for (int qt = wave; qt < KK / 16; qt += 4) {
    int m0 = qt * 16;
    // q fragment straight from global (L2-hot), pre-scaled
    const half_t* qp = QKV + qb + (size_t)(m0 + lm) * C3 + h * HD + hi * 8;
    v16h aq = ldA(qp);
    aq *= hscale;

    // scores S[16][512]
    for (int kt = 0; kt < KK / 16; ++kt) {
      v16h bk = ldA(Kt + (kt * 16 + lm) * HD + hi * 8);
      v8f s = wmma_f16(aq, bk, v8f{});
#pragma unroll
      for (int g = 0; g < 8; ++g)
        Sw[(g + hi * 8) * S_STRIDE + kt * 16 + lm] = s[g];
    }

    // two-pass softmax, row by row (P overlays score region safely:
    // P row r ends before S row r+1 begins)
    float sums[16];
#pragma unroll 1
    for (int r = 0; r < 16; ++r) {
      float v[16];
      float mx = -3.0e38f;
#pragma unroll
      for (int j = 0; j < 16; ++j) {
        v[j] = Sw[r * S_STRIDE + lane + j * 32];
        mx = fmaxf(mx, v[j]);
      }
      for (int off = 16; off > 0; off >>= 1)
        mx = fmaxf(mx, __shfl_xor(mx, off, 32));
      float s = 0.f;
#pragma unroll
      for (int j = 0; j < 16; ++j) {
        v[j] = __expf(v[j] - mx);
        s += v[j];
      }
      for (int off = 16; off > 0; off >>= 1)
        s += __shfl_xor(s, off, 32);
      sums[r] = s;
#pragma unroll
      for (int j = 0; j < 16; ++j)
        Pw[r * P_STRIDE + lane + j * 32] = (half_t)v[j];
    }

    // O = P @ V  (two 16x16 output tiles: d 0..15 and 16..31)
    v8f o0 = {}, o1 = {};
    for (int kc = 0; kc < KK / 32; ++kc) {
      v16h ap = ldA(Pw + lm * P_STRIDE + kc * 32 + hi * 8);
      const half_t* vp = Vs + (size_t)lm * KT_STRIDE + kc * 32 + hi * 8;
      v16h bv0 = ldA(vp);
      v16h bv1 = ldA(vp + 16 * KT_STRIDE);
      o0 = wmma_f16(ap, bv0, o0);
      o1 = wmma_f16(ap, bv1, o1);
    }

    // normalize + store
#pragma unroll
    for (int g = 0; g < 8; ++g) {
      float inv = 1.0f / (hi ? sums[g + 8] : sums[g]);
      int m = m0 + g + hi * 8;
      size_t orow = ((size_t)(b * KK + m)) * CC + h * HD;
      Osel[orow + lm]      = (half_t)(o0[g] * inv);
      Osel[orow + 16 + lm] = (half_t)(o1[g] * inv);
    }
  }
}

// ---------------------------------------------------------------------------
// 8. projection GEMM + bias + scatter into zeroed output
//    out[b][col][idx[m]] = sum_c Osel[b][m][c]*WprojT[col][c] + bproj[col]
// ---------------------------------------------------------------------------
__global__ void __launch_bounds__(32)
proj_scatter_kernel(const half_t* __restrict__ Osel,
                    const half_t* __restrict__ WprojT,
                    const float* __restrict__ bproj,
                    const int* __restrict__ all_idx,
                    float* __restrict__ out) {
  int b = blockIdx.z;
  int m0 = blockIdx.y * 32;
  int n0 = blockIdx.x * 64;
  int lane = threadIdx.x;
  int lm = lane & 15, hi = lane >> 4;

  v8f acc[2][4] = {};
  const half_t* A0 = Osel + ((size_t)(b * KK + m0 + lm)) * CC + hi * 8;
  const half_t* A1 = A0 + 16 * CC;

  for (int c0 = 0; c0 < CC; c0 += 32) {
    v16h a0 = ldA(A0 + c0);
    v16h a1 = ldA(A1 + c0);
    const half_t* Bp = WprojT + (size_t)(n0 + lm) * CC + c0 + hi * 8;
#pragma unroll
    for (int j = 0; j < 4; ++j) {
      v16h bf = ldA(Bp + j * 16 * CC);
      acc[0][j] = wmma_f16(a0, bf, acc[0][j]);
      acc[1][j] = wmma_f16(a1, bf, acc[1][j]);
    }
  }

  int idxs[2][8];
#pragma unroll
  for (int s = 0; s < 2; ++s)
#pragma unroll
    for (int g = 0; g < 8; ++g)
      idxs[s][g] = all_idx[b * KK + m0 + s * 16 + g + hi * 8];

#pragma unroll
  for (int j = 0; j < 4; ++j) {
    int n = n0 + j * 16 + lm;
    float bias = bproj[n];
    float* ocol = out + (size_t)b * CC * NN + (size_t)n * NN;
#pragma unroll
    for (int s = 0; s < 2; ++s)
#pragma unroll
      for (int g = 0; g < 8; ++g)
        ocol[idxs[s][g]] = acc[s][j][g] + bias;
  }
}

// ---------------------------------------------------------------------------
// launch
// ---------------------------------------------------------------------------
extern "C" void kernel_launch(void* const* d_in, const int* in_sizes, int n_in,
                              void* d_out, int out_size, void* d_ws, size_t ws_size,
                              hipStream_t stream) {
  const float* x     = (const float*)d_in[0];
  const float* Wqkv  = (const float*)d_in[1];
  const float* Wproj = (const float*)d_in[2];
  const float* bproj = (const float*)d_in[3];
  float* out = (float*)d_out;

  char* ws = (char*)d_ws;
  float*  norms   = (float*)(ws + 0);            //  819,200 B
  int*    all_idx = (int*)  (ws + 819200);       //   65,536 B
  half_t* WqkvT   = (half_t*)(ws + 884736);      //  393,216 B
  half_t* WprojT  = (half_t*)(ws + 1277952);     //  131,072 B
  half_t* Xsel    = (half_t*)(ws + 1409024);     // 8,388,608 B
  half_t* QKV     = (half_t*)(ws + 9797632);     // 25,165,824 B
  half_t* Osel    = (half_t*)(ws + 34963456);    // 8,388,608 B  (total ~43.4 MB)

  int n4 = (BB * CC * NN) / 4;
  zero_kernel<<<dim3(8192), 256, 0, stream>>>(out, n4);
  norms_kernel<<<dim3(NN / 256, BB), 256, 0, stream>>>(x, norms);
  topk_kernel<<<dim3(NN / 256, BB), 256, 0, stream>>>(norms, all_idx);
  convw_kernel<<<dim3((C3 * CC + CC * CC) / 256), 256, 0, stream>>>(
      Wqkv, Wproj, WqkvT, WprojT);
  gather_kernel<<<dim3(CC / 4, BB), 512, 0, stream>>>(x, all_idx, Xsel);
  qkv_gemm_kernel<<<dim3(C3 / 64, KK / 32, BB), 32, 0, stream>>>(Xsel, WqkvT, QKV);
  attn_kernel<<<dim3(NH, BB), 128, ATTN_LDS_BYTES, stream>>>(QKV, Osel);
  proj_scatter_kernel<<<dim3(CC / 64, KK / 32, BB), 32, 0, stream>>>(
      Osel, WprojT, bproj, all_idx, out);
}
